// CTCA_54743653155443
// MI455X (gfx1250) — compile-verified, matrix operand order checked by reading, not run on hardware
//
#include <hip/hip_runtime.h>
#include <hip/hip_bf16.h>

typedef __attribute__((ext_vector_type(16))) _Float16 v16h;
typedef __attribute__((ext_vector_type(8)))  _Float16 v8h;
typedef __attribute__((ext_vector_type(8)))  float    v8f;

#define NB   8
#define NC   384
#define HWP  128
#define NPIX (HWP * HWP)   /* 16384 */
#define HIN  256

__device__ __forceinline__ v16h pack16(v8h lo, v8h hi) {
    v16h r;
    #pragma unroll
    for (int i = 0; i < 8; i++) { r[i] = lo[i]; r[i + 8] = hi[i]; }
    return r;
}

// ---------------------------------------------------------------- reductions
__device__ __forceinline__ float blockReduceSum(float v) {
    #pragma unroll
    for (int o = 16; o > 0; o >>= 1) v += __shfl_down(v, o, 32);
    __shared__ float red_s[8];
    __shared__ float tot_s;
    int lane = threadIdx.x & 31, w = threadIdx.x >> 5;
    if (lane == 0) red_s[w] = v;
    __syncthreads();
    if (threadIdx.x == 0) {
        float r = 0.f;
        int nw = blockDim.x >> 5;
        for (int i = 0; i < nw; i++) r += red_s[i];
        tot_s = r;
    }
    __syncthreads();
    return tot_s;
}

__device__ __forceinline__ float blockReduceMax(float v) {
    #pragma unroll
    for (int o = 16; o > 0; o >>= 1) v = fmaxf(v, __shfl_down(v, o, 32));
    __shared__ float redm_s[8];
    __shared__ float totm_s;
    int lane = threadIdx.x & 31, w = threadIdx.x >> 5;
    if (lane == 0) redm_s[w] = v;
    __syncthreads();
    if (threadIdx.x == 0) {
        float r = -3.402823466e38f;
        int nw = blockDim.x >> 5;
        for (int i = 0; i < nw; i++) r = fmaxf(r, redm_s[i]);
        totm_s = r;
    }
    __syncthreads();
    return totm_s;
}

// ------------------------------------------------- 1) pool x -> v_f16, invk
// one block per (b,c) row; fused 2x2 avg-pool + sum-of-squares for L2 norm
__global__ void pool_rownorm_kernel(const float* __restrict__ x,
                                    _Float16* __restrict__ v_h,
                                    float* __restrict__ invk) {
    int row = blockIdx.x;                         // b*NC + c
    const float* xr = x + (size_t)row * (HIN * HIN);
    _Float16* vr = v_h + (size_t)row * NPIX;
    float ss = 0.f;
    for (int p = threadIdx.x; p < NPIX; p += blockDim.x) {
        int i = p >> 7, j = p & 127;
        const float* px = xr + (size_t)(i * 2) * HIN + j * 2;
        float val = 0.25f * (px[0] + px[1] + px[HIN] + px[HIN + 1]);
        vr[p] = (_Float16)val;
        ss += val * val;
    }
    ss = blockReduceSum(ss);
    if (threadIdx.x == 0) invk[row] = 1.0f / fmaxf(sqrtf(ss), 1e-12f);
}

// ------------------------------------------------------- 2) q row norms
__global__ void q_rownorm_kernel(const float* __restrict__ q,
                                 float* __restrict__ invq) {
    int row = blockIdx.x;
    const float* qr = q + (size_t)row * NPIX;
    float ss = 0.f;
    for (int p = threadIdx.x; p < NPIX; p += blockDim.x) {
        float v = qr[p];
        ss += v * v;
    }
    ss = blockReduceSum(ss);
    if (threadIdx.x == 0) invq[row] = 1.0f / fmaxf(sqrtf(ss), 1e-12f);
}

// -------------------------- 3) S = (Qn . Kn^T) * temp   [b,384,384]
// norms factored out of the dot product: S = diag(invq) (Qraw.Krawt) diag(invk) * t
// one wave per 16(M) x 64(N) strip: one A fragment feeds 4 WMMAs; B fragments
// are pure f16 loads from the pooled buffer (no per-element VALU in the loop)
__global__ void attn_qkT_kernel(const float* __restrict__ q,
                                const _Float16* __restrict__ v_h,
                                const float* __restrict__ invq,
                                const float* __restrict__ invk,
                                const float* __restrict__ temp,
                                float* __restrict__ S) {
    int b = blockIdx.y;
    int wave = threadIdx.x >> 5;
    int lane = threadIdx.x & 31;
    int strip = blockIdx.x * 4 + wave;            // 0..143
    int m0 = (strip / 6) * 16;
    int n0 = (strip % 6) * 64;
    int half = lane >> 4;
    const float* qp = q + ((size_t)(b * NC + m0 + (lane & 15))) * NPIX + half * 8;
    const _Float16* kp0 = v_h + ((size_t)(b * NC + n0      + (lane & 15))) * NPIX + half * 8;
    const _Float16* kp1 = kp0 + (size_t)16 * NPIX;
    const _Float16* kp2 = kp0 + (size_t)32 * NPIX;
    const _Float16* kp3 = kp0 + (size_t)48 * NPIX;
    v8f acc0 = {}, acc1 = {}, acc2 = {}, acc3 = {};
    for (int kb = 0; kb < NPIX; kb += 32) {
        __builtin_prefetch(qp + kb + 512);
        __builtin_prefetch(kp0 + kb + 1024);
        v8f q0 = *(const v8f*)(qp + kb);
        v8f q1 = *(const v8f*)(qp + kb + 16);
        v16h a;
        #pragma unroll
        for (int i = 0; i < 8; i++) {
            a[i]     = (_Float16)q0[i];
            a[i + 8] = (_Float16)q1[i];
        }
        v16h b0 = pack16(*(const v8h*)(kp0 + kb), *(const v8h*)(kp0 + kb + 16));
        v16h b1 = pack16(*(const v8h*)(kp1 + kb), *(const v8h*)(kp1 + kb + 16));
        v16h b2 = pack16(*(const v8h*)(kp2 + kb), *(const v8h*)(kp2 + kb + 16));
        v16h b3 = pack16(*(const v8h*)(kp3 + kb), *(const v8h*)(kp3 + kb + 16));
        acc0 = __builtin_amdgcn_wmma_f32_16x16x32_f16(false, a, false, b0, (short)0, acc0, false, false);
        acc1 = __builtin_amdgcn_wmma_f32_16x16x32_f16(false, a, false, b1, (short)0, acc1, false, false);
        acc2 = __builtin_amdgcn_wmma_f32_16x16x32_f16(false, a, false, b2, (short)0, acc2, false, false);
        acc3 = __builtin_amdgcn_wmma_f32_16x16x32_f16(false, a, false, b3, (short)0, acc3, false, false);
    }
    // epilogue: apply invq[m] * invk[n] * temperature
    float t = temp[0];
    float fm[8];
    #pragma unroll
    for (int r = 0; r < 8; r++) fm[r] = invq[b * NC + m0 + r + 8 * half] * t;
    v8f accs[4] = { acc0, acc1, acc2, acc3 };
    #pragma unroll
    for (int j = 0; j < 4; j++) {
        int gn = n0 + 16 * j + (lane & 15);
        float sk = invk[b * NC + gn];
        #pragma unroll
        for (int r = 0; r < 8; r++) {
            int gm = m0 + r + 8 * half;
            S[((size_t)(b * NC) + gm) * NC + gn] = accs[j][r] * fm[r] * sk;
        }
    }
}

// ------------------- 4) column softmax over cq, write transposed At[b][d][cq]
__global__ void softmax_col_kernel(const float* __restrict__ S,
                                   _Float16* __restrict__ At) {
    int b = blockIdx.y;
    int d = blockIdx.x;                           // 0..383
    const float* col = S + (size_t)b * NC * NC + d;   // stride NC
    float vals[3];
    float lmax = -3.402823466e38f;
    #pragma unroll
    for (int i = 0; i < 3; i++) {
        int cq = threadIdx.x + i * 128;
        float v = col[(size_t)cq * NC];
        vals[i] = v;
        lmax = fmaxf(lmax, v);
    }
    float m = blockReduceMax(lmax);
    float ls = 0.f;
    #pragma unroll
    for (int i = 0; i < 3; i++) {
        vals[i] = __expf(vals[i] - m);
        ls += vals[i];
    }
    float s = blockReduceSum(ls);
    float inv = 1.f / s;
    #pragma unroll
    for (int i = 0; i < 3; i++) {
        int cq = threadIdx.x + i * 128;
        At[((size_t)b * NC + d) * NC + cq] = (_Float16)(vals[i] * inv);
    }
}

// ------------------------- 5) T = P . A   (uses At rows as B-fragments), f16
__global__ void proj_attn_kernel(const float* __restrict__ P,
                                 const _Float16* __restrict__ At,
                                 _Float16* __restrict__ T) {
    int b = blockIdx.y;
    int wave = threadIdx.x >> 5;
    int lane = threadIdx.x & 31;
    int tile = blockIdx.x * 4 + wave;             // 0..575
    int m0 = (tile / 24) * 16;
    int n0 = (tile % 24) * 16;
    int half = lane >> 4;
    const float*    pp = P  + (size_t)(m0 + (lane & 15)) * NC + half * 8;
    const _Float16* ap = At + ((size_t)b * NC + n0 + (lane & 15)) * NC + half * 8;
    v8f acc = {};
    for (int kb = 0; kb < NC; kb += 32) {
        v8f p0 = *(const v8f*)(pp + kb);
        v8f p1 = *(const v8f*)(pp + kb + 16);
        v16h a;
        #pragma unroll
        for (int i = 0; i < 8; i++) {
            a[i]     = (_Float16)p0[i];
            a[i + 8] = (_Float16)p1[i];
        }
        v16h bf = pack16(*(const v8h*)(ap + kb), *(const v8h*)(ap + kb + 16));
        acc = __builtin_amdgcn_wmma_f32_16x16x32_f16(false, a, false, bf,
                                                     (short)0, acc, false, false);
    }
    #pragma unroll
    for (int r = 0; r < 8; r++) {
        int gm = m0 + r + 8 * half;
        int gn = n0 + (lane & 15);
        T[((size_t)b * NC + gm) * NC + gn] = (_Float16)acc[r];
    }
}

// ---------------------- 6) Y = T . V  with LDS-transposed V tiles [b,384,16384]
__global__ void out_gemm_kernel(const _Float16* __restrict__ T,
                                const _Float16* __restrict__ v_h,
                                float* __restrict__ Y) {
    __shared__ _Float16 vt[64][40];               // [j][d], padded stride
    int b  = blockIdx.z;
    int n0 = blockIdx.x * 64;
    int m0 = blockIdx.y * 64;
    int tid = threadIdx.x;
    int wave = tid >> 5, lane = tid & 31, half = lane >> 4;
    int wm = (wave & 3) * 16;                     // wave grid 4(M) x 2(N)
    int wn = (wave >> 2) * 32;
    const _Float16* tp = T + ((size_t)b * NC + m0 + wm + (lane & 15)) * NC + half * 8;
    const _Float16* vb = v_h + (size_t)b * NC * NPIX;
    v8f acc0 = {}, acc1 = {};
    for (int kb = 0; kb < NC; kb += 32) {
        // stage V[kb..kb+32)[n0..n0+64) transposed into LDS
        int d  = tid >> 3;                        // 0..31
        int jc = (tid & 7) * 8;                   // 0..56
        v8h vv = *(const v8h*)(vb + (size_t)(kb + d) * NPIX + n0 + jc);
        #pragma unroll
        for (int i = 0; i < 8; i++) vt[jc + i][d] = vv[i];
        __syncthreads();

        v16h a = pack16(*(const v8h*)(tp + kb), *(const v8h*)(tp + kb + 16));

        const _Float16* r0 = &vt[wn + (lane & 15)][half * 8];
        const _Float16* r1 = &vt[wn + 16 + (lane & 15)][half * 8];
        v16h b0 = pack16(*(const v8h*)r0, *(const v8h*)(r0 + 16));
        v16h b1 = pack16(*(const v8h*)r1, *(const v8h*)(r1 + 16));
        acc0 = __builtin_amdgcn_wmma_f32_16x16x32_f16(false, a, false, b0,
                                                      (short)0, acc0, false, false);
        acc1 = __builtin_amdgcn_wmma_f32_16x16x32_f16(false, a, false, b1,
                                                      (short)0, acc1, false, false);
        __syncthreads();
    }
    #pragma unroll
    for (int r = 0; r < 8; r++) {
        int gm = m0 + wm + r + 8 * half;
        size_t base = ((size_t)b * NC + gm) * NPIX + n0;
        Y[base + wn + (lane & 15)]      = acc0[r];
        Y[base + wn + 16 + (lane & 15)] = acc1[r];
    }
}

// --------------------------------------------------------------------- launch
extern "C" void kernel_launch(void* const* d_in, const int* in_sizes, int n_in,
                              void* d_out, int out_size, void* d_ws, size_t ws_size,
                              hipStream_t stream) {
    const float* x    = (const float*)d_in[0];   // [8,384,256,256]
    const float* q    = (const float*)d_in[1];   // [8,384,128,128]
    const float* temp = (const float*)d_in[2];   // [1,1,1]
    const float* pw   = (const float*)d_in[3];   // [384,384]
    float* out = (float*)d_out;                  // [8,384,128,128]

    // workspace layout (~105 MiB)
    char* w = (char*)d_ws;
    const size_t SZ_V = (size_t)NB * NC * NPIX * sizeof(_Float16);   // 100,663,296
    _Float16* v_h  = (_Float16*)w;
    float*    invk = (float*)(w + SZ_V);
    float*    invq = invk + (size_t)NB * NC;
    float*    S    = (float*)(w + SZ_V + 2 * (size_t)NB * NC * sizeof(float));
    _Float16* At   = (_Float16*)((char*)S + (size_t)NB * NC * NC * sizeof(float));
    _Float16* T    = (_Float16*)((char*)At + (size_t)NB * NC * NC * sizeof(_Float16));

    // 1) pool + k row norms
    pool_rownorm_kernel<<<dim3(NB * NC), dim3(256), 0, stream>>>(x, v_h, invk);
    // 2) q row norms
    q_rownorm_kernel<<<dim3(NB * NC), dim3(256), 0, stream>>>(q, invq);
    // 3) S = Qn Kn^T * temperature  (144 strips of 16x64 / batch, 4 waves / block)
    attn_qkT_kernel<<<dim3(36, NB), dim3(128), 0, stream>>>(q, v_h, invq, invk, temp, S);
    // 4) column softmax -> At (transposed probabilities)
    softmax_col_kernel<<<dim3(NC, NB), dim3(128), 0, stream>>>(S, At);
    // 5) T = proj_w @ A
    proj_attn_kernel<<<dim3(144, NB), dim3(128), 0, stream>>>(pw, At, T);
    // 6) Y = T @ V
    out_gemm_kernel<<<dim3(NPIX / 64, NC / 64, NB), dim3(256), 0, stream>>>(T, v_h, out);
}
